// Encoder_44049184588235
// MI455X (gfx1250) — matrix operand order repaired and checked by reading, not compiled, not run on hardware
//
#include <hip/hip_runtime.h>
#include <hip/hip_bf16.h>

typedef _Float16 f16;
typedef __attribute__((ext_vector_type(8)))  _Float16 v8h;
typedef __attribute__((ext_vector_type(16))) _Float16 v16h;
typedef __attribute__((ext_vector_type(8)))  float    v8f;

#define S_IN   44
#define HID    128
#define DDIM   64
#define NVEC   16
#define NHID   16
#define VCIN   17
#define RBFD   16
#define EFD    16
#define KPAD   128   // 96 scalar + 16 vnorm + 16 zero pad
#define TILE   16
#define WPB    4     // waves per block (blockDim = 128)

// ---------------- utility kernels ----------------

__global__ void zero_kernel(float* p, int n) {
  int i = blockIdx.x * blockDim.x + threadIdx.x;
  if (i < n) p[i] = 0.0f;
}

__global__ void copy_kernel(const float* __restrict__ src, float* __restrict__ dst, int n) {
  int i = blockIdx.x * blockDim.x + threadIdx.x;
  if (i < n) dst[i] = src[i];
}

__global__ void deg_kernel(const int* __restrict__ eidx, float* __restrict__ deg, int E) {
  int e = blockIdx.x * blockDim.x + threadIdx.x;
  if (e < E) atomicAdd(&deg[eidx[E + e]], 1.0f);
}

// s = relu(x@W1+b1)@W2+b2 : one node per block (128 threads)
__global__ __launch_bounds__(HID)
void embed_kernel(const float* __restrict__ x,
                  const float* __restrict__ W1, const float* __restrict__ b1,
                  const float* __restrict__ W2, const float* __restrict__ b2,
                  float* __restrict__ out) {
  __shared__ float xs[S_IN];
  __shared__ float hs[HID];
  const int n = blockIdx.x;
  const int t = threadIdx.x;
  if (t < S_IN) xs[t] = x[(size_t)n * S_IN + t];
  __syncthreads();
  float h = b1[t];
  #pragma unroll 4
  for (int i = 0; i < S_IN; ++i) h += xs[i] * W1[i * HID + t];
  hs[t] = fmaxf(h, 0.0f);
  __syncthreads();
  if (t < DDIM) {
    float o = b2[t];
    #pragma unroll 4
    for (int j = 0; j < HID; ++j) o += hs[j] * W2[j * DDIM + t];
    out[(size_t)n * DDIM + t] = o;
  }
}

// residual mean-aggregate update: s += s_agg/deg ; v += v_agg/deg
__global__ void update_kernel(float* __restrict__ s, float* __restrict__ v,
                              const float* __restrict__ s_agg,
                              const float* __restrict__ v_agg,
                              const float* __restrict__ deg, int N) {
  int i = blockIdx.x * blockDim.x + threadIdx.x;
  int tot = N * (DDIM + NVEC * 3);
  if (i >= tot) return;
  if (i < N * DDIM) {
    float dg = fmaxf(deg[i / DDIM], 1.0f);
    s[i] += s_agg[i] / dg;
  } else {
    int j = i - N * DDIM;
    float dg = fmaxf(deg[j / (NVEC * 3)], 1.0f);
    v[j] += v_agg[j] / dg;
  }
}

// ---------------- GVP message kernel (WMMA) ----------------
// One wave per 16-edge tile. Per-edge feature row (K=112, padded 128):
//   [ s[src] (64) | rbf(d) (16) | edge_feats (16) | vnorm (16) | 0 pad (16) ]
// sm   = relu(row @ Ws + bs)        -> 16x64 via v_wmma_f32_16x16x32_f16 (4 Ksteps x 4 Ntiles)
// gate = sigmoid(sm @ Wg + bg)      -> 16x16 via 2 WMMAs
// Vh / Vmu vector lifts in f32 VALU; scatter via global_atomic_add_f32.
__global__ __launch_bounds__(WPB * 32)
void msg_kernel(const float* __restrict__ s, const float* __restrict__ vcur,
                const float* __restrict__ ef, const float* __restrict__ xdiff,
                const float* __restrict__ dvals, const int* __restrict__ eidx,
                const float* __restrict__ Wh, const float* __restrict__ Ws,
                const float* __restrict__ bs, const float* __restrict__ Wmu,
                const float* __restrict__ Wg, const float* __restrict__ bg,
                float* __restrict__ s_agg, float* __restrict__ v_agg, int E) {
  __shared__ f16   WsT[DDIM][KPAD];       // [n][k], K padded with zeros
  __shared__ f16   WgT[NVEC][DDIM];       // [n][k]
  __shared__ float WhS[VCIN][NHID];
  __shared__ float WmuS[NHID][NVEC];
  __shared__ float bsS[DDIM];
  __shared__ float bgS[NVEC];
  __shared__ float VhS[WPB][TILE][NHID * 3];
  __shared__ f16   sA[WPB][TILE][KPAD];   // A tile (f16)
  __shared__ f16   smA[WPB][TILE][DDIM];  // sm tile (f16) for gate GEMM
  __shared__ float gateS[WPB][TILE][NVEC];
  __shared__ int   srcS[WPB][TILE];
  __shared__ int   dstS[WPB][TILE];

  const int tid = threadIdx.x;

  // cooperative weight staging
  for (int i = tid; i < DDIM * KPAD; i += WPB * 32) {
    int n = i >> 7, k = i & 127;
    WsT[n][k] = (k < 112) ? (f16)Ws[k * DDIM + n] : (f16)0.0f;
  }
  for (int i = tid; i < NVEC * DDIM; i += WPB * 32) {
    int n = i >> 6, k = i & 63;
    WgT[n][k] = (f16)Wg[k * NVEC + n];
  }
  for (int i = tid; i < VCIN * NHID; i += WPB * 32) WhS[i >> 4][i & 15] = Wh[i];
  for (int i = tid; i < NHID * NVEC; i += WPB * 32) WmuS[i >> 4][i & 15] = Wmu[i];
  if (tid < DDIM) bsS[tid] = bs[tid];
  if (tid < NVEC) bgS[tid] = bg[tid];

  const int wv = tid >> 5, lane = tid & 31;
  const int m = lane & 15;        // edge-in-tile / N-col depending on phase
  const int hi = lane >> 4;       // lane half
  const int e0 = (blockIdx.x * WPB + wv) * TILE;

  if (lane < TILE) {
    int e = (e0 + lane < E) ? (e0 + lane) : (E - 1);
    srcS[wv][lane] = eidx[e];
    dstS[wv][lane] = eidx[E + e];
  }
  __syncthreads();

  // ---- Vh = Wh^T * v_in  and vnorm (f32 VALU) ----
  {
    int e = (e0 + m < E) ? (e0 + m) : (E - 1);
    int sn = srcS[wv][m];
    int hb = hi * 8;
    float acc[8][3];
    #pragma unroll
    for (int hh = 0; hh < 8; ++hh) { acc[hh][0] = acc[hh][1] = acc[hh][2] = 0.0f; }
    for (int c = 0; c < VCIN; ++c) {
      float vx, vy, vz;
      if (c < NVEC) {
        size_t b = (size_t)sn * (NVEC * 3) + c * 3;
        vx = vcur[b]; vy = vcur[b + 1]; vz = vcur[b + 2];
      } else {
        size_t b = (size_t)e * 3;
        vx = xdiff[b]; vy = xdiff[b + 1]; vz = xdiff[b + 2];
      }
      #pragma unroll
      for (int hh = 0; hh < 8; ++hh) {
        float w = WhS[c][hb + hh];
        acc[hh][0] += vx * w; acc[hh][1] += vy * w; acc[hh][2] += vz * w;
      }
    }
    #pragma unroll
    for (int hh = 0; hh < 8; ++hh) {
      int h = hb + hh;
      VhS[wv][m][h * 3 + 0] = acc[hh][0];
      VhS[wv][m][h * 3 + 1] = acc[hh][1];
      VhS[wv][m][h * 3 + 2] = acc[hh][2];
      float vn = sqrtf(acc[hh][0] * acc[hh][0] + acc[hh][1] * acc[hh][1] +
                       acc[hh][2] * acc[hh][2] + 1e-8f);
      sA[wv][m][96 + h] = (f16)vn;
    }
  }
  // ---- fill scalar part of A tile ----
  {
    int e = (e0 + m < E) ? (e0 + m) : (E - 1);
    int sn = srcS[wv][m];
    if (hi == 0) {
      for (int j = 0; j < 32; ++j) sA[wv][m][j] = (f16)s[(size_t)sn * DDIM + j];
      float dd = dvals[e];
      #pragma unroll
      for (int j = 0; j < RBFD; ++j) {
        float mu = (10.0f / 15.0f) * (float)j;       // linspace(0, DMAX, 16)
        float t = (dd - mu) * (16.0f / 10.0f);       // /sigma, sigma = dmax/dim
        sA[wv][m][64 + j] = (f16)__expf(-t * t);
      }
      for (int j = 0; j < EFD; ++j) sA[wv][m][80 + j] = (f16)ef[(size_t)e * EFD + j];
    } else {
      for (int j = 0; j < 32; ++j) sA[wv][m][32 + j] = (f16)s[(size_t)sn * DDIM + 32 + j];
      for (int j = 0; j < 16; ++j) sA[wv][m][112 + j] = (f16)0.0f;
    }
  }
  __syncthreads();

  // ---- sm = relu(A @ Ws + bs) via WMMA f32_16x16x32_f16 ----
  const int koffA = hi * 16;   // A frag: 16 contiguous K starting at koffA
  const int koffB = hi * 16;   // B frag: 16 contiguous K starting at koffB
  v8f cacc[4];
  #pragma unroll
  for (int nt = 0; nt < 4; ++nt)
    #pragma unroll
    for (int r = 0; r < 8; ++r) cacc[nt][r] = 0.0f;

  #pragma unroll
  for (int ks = 0; ks < 4; ++ks) {
    v8h alo = *(const v8h*)&sA[wv][m][ks * 32 + koffA];
    v8h ahi = *(const v8h*)&sA[wv][m][ks * 32 + 8 + koffA];
    v16h afrag;
    #pragma unroll
    for (int i = 0; i < 8; ++i) { afrag[i] = alo[i]; afrag[8 + i] = ahi[i]; }
    #pragma unroll
    for (int nt = 0; nt < 4; ++nt) {
      v16h bfrag = *(const v16h*)&WsT[nt * 16 + m][ks * 32 + koffB];
      cacc[nt] = __builtin_amdgcn_wmma_f32_16x16x32_f16(
          false, afrag, false, bfrag, (short)0, cacc[nt], false, false);
    }
  }

  // bias + relu (C layout: row = hi*8+r, col = nt*16+m); stage f16 + scatter
  #pragma unroll
  for (int nt = 0; nt < 4; ++nt) {
    int col = nt * 16 + m;
    float bias = bsS[col];
    #pragma unroll
    for (int r = 0; r < 8; ++r) {
      int mr = hi * 8 + r;
      float val = fmaxf(cacc[nt][r] + bias, 0.0f);
      smA[wv][mr][col] = (f16)val;
      if (e0 + mr < E)
        atomicAdd(&s_agg[(size_t)dstS[wv][mr] * DDIM + col], val);
    }
  }
  __syncthreads();

  // ---- gate = sigmoid(sm @ Wg + bg) via 2 WMMAs (K=64) ----
  v8f cg;
  #pragma unroll
  for (int r = 0; r < 8; ++r) cg[r] = 0.0f;
  #pragma unroll
  for (int ks = 0; ks < 2; ++ks) {
    v8h alo = *(const v8h*)&smA[wv][m][ks * 32 + koffA];
    v8h ahi = *(const v8h*)&smA[wv][m][ks * 32 + 8 + koffA];
    v16h afrag;
    #pragma unroll
    for (int i = 0; i < 8; ++i) { afrag[i] = alo[i]; afrag[8 + i] = ahi[i]; }
    v16h bfrag = *(const v16h*)&WgT[m][ks * 32 + koffB];
    cg = __builtin_amdgcn_wmma_f32_16x16x32_f16(
        false, afrag, false, bfrag, (short)0, cg, false, false);
  }
  #pragma unroll
  for (int r = 0; r < 8; ++r) {
    int mr = hi * 8 + r;
    float xg = cg[r] + bgS[m];
    gateS[wv][mr][m] = 1.0f / (1.0f + __expf(-xg));
  }
  __syncthreads();

  // ---- Vmu = Wmu^T * Vh ; vm = Vmu*gate ; scatter ----
  {
    int ob = hi * 8;
    int dn = dstS[wv][m];
    bool valid = (e0 + m) < E;
    #pragma unroll
    for (int oo = 0; oo < 8; ++oo) {
      int o = ob + oo;
      float ax = 0.0f, ay = 0.0f, az = 0.0f;
      #pragma unroll
      for (int h = 0; h < NHID; ++h) {
        float w = WmuS[h][o];
        ax += VhS[wv][m][h * 3 + 0] * w;
        ay += VhS[wv][m][h * 3 + 1] * w;
        az += VhS[wv][m][h * 3 + 2] * w;
      }
      float g = gateS[wv][m][o];
      if (valid) {
        size_t b = (size_t)dn * (NVEC * 3) + o * 3;
        atomicAdd(&v_agg[b + 0], ax * g);
        atomicAdd(&v_agg[b + 1], ay * g);
        atomicAdd(&v_agg[b + 2], az * g);
      }
    }
  }
}

// ---------------- host launcher ----------------

extern "C" void kernel_launch(void* const* d_in, const int* in_sizes, int n_in,
                              void* d_out, int out_size, void* d_ws, size_t ws_size,
                              hipStream_t stream) {
  const float* scalar_feats = (const float*)d_in[0];
  // d_in[1] = coord_feats (unused by reference math)
  const float* vector_feats = (const float*)d_in[2];
  const float* edge_feats   = (const float*)d_in[3];
  const float* x_diff       = (const float*)d_in[4];
  const float* dvals        = (const float*)d_in[5];
  const int*   eidx         = (const int*)d_in[6];
  const float* W1  = (const float*)d_in[7];
  const float* b1  = (const float*)d_in[8];
  const float* W2  = (const float*)d_in[9];
  const float* b2  = (const float*)d_in[10];
  const float* Wh  = (const float*)d_in[11];
  const float* Ws  = (const float*)d_in[12];
  const float* bs  = (const float*)d_in[13];
  const float* Wmu = (const float*)d_in[14];
  const float* Wg  = (const float*)d_in[15];
  const float* bg  = (const float*)d_in[16];

  const int N = in_sizes[0] / S_IN;
  const int E = in_sizes[5];

  float* s     = (float*)d_out;                 // node scalar state lives in d_out
  float* v_ws  = (float*)d_ws;                  // N*48
  float* s_agg = v_ws + (size_t)N * (NVEC * 3); // N*64  (s_agg,v_agg contiguous)
  float* v_agg = s_agg + (size_t)N * DDIM;      // N*48
  float* deg   = v_agg + (size_t)N * (NVEC * 3);// N

  const int aggN = N * (DDIM + NVEC * 3);

  zero_kernel<<<(N + 255) / 256, 256, 0, stream>>>(deg, N);
  deg_kernel<<<(E + 255) / 256, 256, 0, stream>>>(eidx, deg, E);
  copy_kernel<<<(N * NVEC * 3 + 255) / 256, 256, 0, stream>>>(vector_feats, v_ws, N * NVEC * 3);
  embed_kernel<<<N, HID, 0, stream>>>(scalar_feats, W1, b1, W2, b2, s);

  const int mblocks = (E + WPB * TILE - 1) / (WPB * TILE);
  for (int l = 0; l < 3; ++l) {
    zero_kernel<<<(aggN + 255) / 256, 256, 0, stream>>>(s_agg, aggN);
    msg_kernel<<<mblocks, WPB * 32, 0, stream>>>(
        s, v_ws, edge_feats, x_diff, dvals, eidx,
        Wh + (size_t)l * VCIN * NHID,
        Ws + (size_t)l * 112 * DDIM,
        bs + (size_t)l * DDIM,
        Wmu + (size_t)l * NHID * NVEC,
        Wg + (size_t)l * DDIM * NVEC,
        bg + (size_t)l * NVEC,
        s_agg, v_agg, E);
    update_kernel<<<(aggN + 255) / 256, 256, 0, stream>>>(s, v_ws, s_agg, v_agg, deg, N);
  }
}
